// GraphEncoder_54546084659488
// MI455X (gfx1250) — matrix operand order repaired and checked by reading, not compiled, not run on hardware
//
#include <hip/hip_runtime.h>
#include <hip/hip_bf16.h>

#define B_ 128
#define T_ 512
#define N_ 256
#define H_ 1024

typedef __attribute__((ext_vector_type(16))) __bf16 v16bf;
typedef __attribute__((ext_vector_type(8)))  __bf16 v8bf;
typedef __attribute__((ext_vector_type(8)))  float  v8f;

static __device__ __forceinline__ float sigmoidf_(float x) {
    return 1.0f / (1.0f + __expf(-x));
}

// ---------------------------------------------------------------------------
// Weight prep: fp32 [K,N] row-major -> bf16 [N,K] (transposed) so GEMM tiles
// are contiguous per output-column row and can be async-copied to LDS.
// ---------------------------------------------------------------------------
__global__ void k_cvt_t(__bf16* __restrict__ dst, const float* __restrict__ src,
                        int K, int N) {
    int total = K * N;
    for (int i = blockIdx.x * blockDim.x + threadIdx.x; i < total;
         i += gridDim.x * blockDim.x) {
        int n = i / K, k = i - n * K;
        dst[i] = (__bf16)src[(long)k * N + n];
    }
}

// dst[N][K] = bf16(Wr - scale*Wl)^T   (folded SAGE weight, transposed)
__global__ void k_combine_t(__bf16* __restrict__ dst, const float* __restrict__ wr,
                            const float* __restrict__ wl, float scale, int K, int N) {
    int total = K * N;
    for (int i = blockIdx.x * blockDim.x + threadIdx.x; i < total;
         i += gridDim.x * blockDim.x) {
        int n = i / K, k = i - n * K;
        long s = (long)k * N + n;
        dst[i] = (__bf16)(wr[s] - scale * wl[s]);
    }
}

// x [B,T,N] fp32 -> xT [B,N,T] bf16
__global__ void k_xT(__bf16* __restrict__ xT, const float* __restrict__ x) {
    const long total = (long)B_ * T_ * N_;
    for (long i = (long)blockIdx.x * blockDim.x + threadIdx.x; i < total;
         i += (long)gridDim.x * blockDim.x) {
        int  t = (int)(i % T_);
        long r = i / T_;
        int  n = (int)(r % N_);
        int  b = (int)(r / N_);
        xT[i] = (__bf16)x[((long)b * T_ + t) * N_ + n];
    }
}

// s1[b*T+t] = (sum_n x[b,t,n]) / (N-1), bf16. One wave per row.
__global__ void k_rowsum_x(__bf16* __restrict__ s1, const float* __restrict__ x) {
    int row  = blockIdx.x * (blockDim.x >> 5) + (threadIdx.x >> 5);
    int lane = threadIdx.x & 31;
    if (row >= B_ * T_) return;
    const float* p = x + (long)row * N_;
    float s = 0.f;
    for (int i = lane; i < N_; i += 32) s += p[i];
    #pragma unroll
    for (int off = 16; off; off >>= 1) s += __shfl_xor(s, off, 32);
    if (lane == 0) s1[row] = (__bf16)(s * (1.0f / (N_ - 1)));
}

// s2[b*H+h] = (sum_n h1[b,n,h]) / (N-1), bf16. One thread per (b,h).
__global__ void k_colsum_h(__bf16* __restrict__ s2, const __bf16* __restrict__ h) {
    int i = blockIdx.x * blockDim.x + threadIdx.x;
    if (i >= B_ * H_) return;
    int hc = i % H_, b = i / H_;
    const __bf16* p = h + (long)b * N_ * H_ + hc;
    float s = 0.f;
    for (int n = 0; n < N_; ++n) s += (float)p[(long)n * H_];
    s2[i] = (__bf16)(s * (1.0f / (N_ - 1)));
}

// ---------------------------------------------------------------------------
// bf16 WMMA GEMM:  C = act( A[M,K] @ Wt[Nn,K]^T + bias )
//   A  bf16 row-major, ld = K (contiguous)
//   Wt bf16 row-major [Nn,K] (pre-transposed weight), ld = K
// Block tile 128x128, 8 waves (2 M x 4 N), each wave 4x2 tiles of
// v_wmma_f32_16x16x32_bf16, K-step 32. Both LDS tiles are filled with
// GLOBAL_LOAD_ASYNC_TO_LDS_B128 (ASYNCcnt), fenced by s_wait_asynccnt
// before the workgroup barrier. M,Nn multiples of 128, K of 32.
// ---------------------------------------------------------------------------
template <bool SIG, bool TRANS, bool OUTF32>
__global__ __launch_bounds__(256)
void k_gemm(const __bf16* __restrict__ A, long strideA,
            const __bf16* __restrict__ Wt,
            const float* __restrict__ bias, long strideBias,
            void* __restrict__ Cv, long strideC,
            int M, int Nn, int K)
{
    __shared__ alignas(32) __bf16 sA[128 * 32];   // [row][k]
    __shared__ alignas(32) __bf16 sW[128 * 32];   // [n][k]

    const int tid  = threadIdx.x;
    const int b    = blockIdx.z;
    const int row0 = blockIdx.y * 128;
    const int col0 = blockIdx.x * 128;
    A += (long)b * strideA;

    const int wave = tid >> 5, lane = tid & 31;
    const int wm = wave & 1, wn = wave >> 1;       // wave grid 2(M) x 4(N)
    const int lrow = lane & 15, kh = lane >> 4;    // half-lane select

    v8f acc[4][2];
    #pragma unroll
    for (int mt = 0; mt < 4; ++mt)
        #pragma unroll
        for (int nt = 0; nt < 2; ++nt)
            #pragma unroll
            for (int r = 0; r < 8; ++r) acc[mt][nt][r] = 0.f;

    // Per-thread 16B chunk assignment inside a 128x32 bf16 tile:
    // e in [0,512): row = e>>2, seg = e&3 (8 bf16 = 16 bytes per chunk).
    const int nkt = K >> 5;
    for (int kt = 0; kt < nkt; ++kt) {
        __syncthreads();   // previous tile fully consumed before overwrite
        #pragma unroll
        for (int i = 0; i < 2; ++i) {
            int e  = tid + 256 * i;
            int ar = e >> 2, as = e & 3;
            const __bf16* ga = A  + (long)(row0 + ar) * K + kt * 32 + as * 8;
            const __bf16* gw = Wt + (long)(col0 + ar) * K + kt * 32 + as * 8;
            unsigned la = (unsigned)(uintptr_t)&sA[ar * 32 + as * 8];
            unsigned lw = (unsigned)(uintptr_t)&sW[ar * 32 + as * 8];
            asm volatile("global_load_async_to_lds_b128 %0, %1, off"
                         :: "v"(la), "v"((unsigned long long)(uintptr_t)ga)
                         : "memory");
            asm volatile("global_load_async_to_lds_b128 %0, %1, off"
                         :: "v"(lw), "v"((unsigned long long)(uintptr_t)gw)
                         : "memory");
        }
        asm volatile("s_wait_asynccnt 0x0" ::: "memory");
        __syncthreads();

        // ---- fragments: contiguous k = 16*(lane>>4) + s, same for A and B ----
        v16bf af[4], bfr[2];
        #pragma unroll
        for (int mt = 0; mt < 4; ++mt) {
            int r = wm * 64 + mt * 16 + lrow;
            af[mt] = *(const v16bf*)&sA[r * 32 + kh * 16];             // K 0..15 / 16..31
        }
        #pragma unroll
        for (int nt = 0; nt < 2; ++nt) {
            int c = wn * 32 + nt * 16 + lrow;
            bfr[nt] = *(const v16bf*)&sW[c * 32 + kh * 16];            // K 0..15 / 16..31
        }

        #pragma unroll
        for (int mt = 0; mt < 4; ++mt)
            #pragma unroll
            for (int nt = 0; nt < 2; ++nt)
                acc[mt][nt] = __builtin_amdgcn_wmma_f32_16x16x32_bf16(
                    false, af[mt], false, bfr[nt], (short)0, acc[mt][nt], false, false);
    }

    // ---- epilogue: C/D layout is row = r + 8*(lane>>4), col = lane&15 ----
    const float* bp = bias ? bias + (long)b * strideBias : nullptr;
    #pragma unroll
    for (int mt = 0; mt < 4; ++mt) {
        #pragma unroll
        for (int nt = 0; nt < 2; ++nt) {
            int   colg = col0 + wn * 32 + nt * 16 + lrow;
            float bv   = bp ? bp[colg] : 0.f;
            #pragma unroll
            for (int r = 0; r < 8; ++r) {
                int   rowg = row0 + wm * 64 + mt * 16 + kh * 8 + r;
                float v    = acc[mt][nt][r] + bv;
                if (SIG) v = sigmoidf_(v);
                if (OUTF32) {
                    ((float*)Cv)[(long)b * strideC + (long)rowg * Nn + colg] = v;
                } else {
                    __bf16* C = (__bf16*)Cv;
                    if (TRANS) C[(long)b * strideC + (long)colg * M + rowg] = (__bf16)v;
                    else       C[(long)b * strideC + (long)rowg * Nn + colg] = (__bf16)v;
                }
            }
        }
    }
}

// ---------------------------------------------------------------------------
extern "C" void kernel_launch(void* const* d_in, const int* in_sizes, int n_in,
                              void* d_out, int out_size, void* d_ws, size_t ws_size,
                              hipStream_t stream)
{
    const float* x   = (const float*)d_in[0];
    const float* Wl1 = (const float*)d_in[1];
    const float* Wr1 = (const float*)d_in[2];
    const float* b1  = (const float*)d_in[3];
    const float* Wl2 = (const float*)d_in[4];
    const float* Wr2 = (const float*)d_in[5];
    const float* b2  = (const float*)d_in[6];
    const float* W1a = (const float*)d_in[7];
    const float* W1b = (const float*)d_in[8];
    const float* W2a = (const float*)d_in[9];
    const float* W2b = (const float*)d_in[10];
    (void)in_sizes; (void)n_in; (void)out_size; (void)ws_size;

    char*  ws  = (char*)d_ws;
    size_t off = 0;
    auto alloc = [&](size_t bytes) -> void* {
        void* p = ws + off;
        off += (bytes + 255) & ~(size_t)255;
        return p;
    };

    __bf16* xT    = (__bf16*)alloc((size_t)B_ * N_ * T_ * 2);  // later reused as pT
    __bf16* h1    = (__bf16*)alloc((size_t)B_ * N_ * H_ * 2);  // later reused as t1
    __bf16* h2    = (__bf16*)alloc((size_t)B_ * N_ * H_ * 2);  // later reused as t2
    __bf16* s1    = (__bf16*)alloc((size_t)B_ * T_ * 2);
    __bf16* s2    = (__bf16*)alloc((size_t)B_ * H_ * 2);
    float*  bias1 = (float*)alloc((size_t)B_ * H_ * 4);
    float*  bias2 = (float*)alloc((size_t)B_ * H_ * 4);
    // transposed bf16 weights [N,K]
    __bf16* Wl1t  = (__bf16*)alloc((size_t)T_ * H_ * 2);
    __bf16* W1ct  = (__bf16*)alloc((size_t)T_ * H_ * 2);
    __bf16* Wl2t  = (__bf16*)alloc((size_t)H_ * H_ * 2);
    __bf16* W2ct  = (__bf16*)alloc((size_t)H_ * H_ * 2);
    __bf16* W1at  = (__bf16*)alloc((size_t)H_ * H_ * 2);
    __bf16* W1bt  = (__bf16*)alloc((size_t)H_ * T_ * 2);
    __bf16* W2at  = (__bf16*)alloc((size_t)N_ * N_ * 2);
    __bf16* W2bt  = (__bf16*)alloc((size_t)N_ * H_ * 2);

    const float inv = 1.0f / (float)(N_ - 1);

    // ---- weight prep (fp32 -> bf16, transpose, SAGE weight folding) ----
    k_cvt_t    <<<512, 256, 0, stream>>>(Wl1t, Wl1, T_, H_);
    k_combine_t<<<512, 256, 0, stream>>>(W1ct, Wr1, Wl1, inv, T_, H_);
    k_cvt_t    <<<512, 256, 0, stream>>>(Wl2t, Wl2, H_, H_);
    k_combine_t<<<512, 256, 0, stream>>>(W2ct, Wr2, Wl2, inv, H_, H_);
    k_cvt_t    <<<512, 256, 0, stream>>>(W1at, W1a, H_, H_);
    k_cvt_t    <<<512, 256, 0, stream>>>(W1bt, W1b, H_, T_);
    k_cvt_t    <<<128, 256, 0, stream>>>(W2at, W2a, N_, N_);
    k_cvt_t    <<<256, 256, 0, stream>>>(W2bt, W2b, N_, H_);

    // ---- activation prep ----
    k_xT      <<<4096, 256, 0, stream>>>(xT, x);
    k_rowsum_x<<<(B_ * T_) / 8, 256, 0, stream>>>(s1, x);

    // bias1[b,:] = s1[b,:] @ Wl1 + b1     (M=B, N=H, K=T, fp32 out)
    k_gemm<false, false, true><<<dim3(H_ / 128, 1, 1), 256, 0, stream>>>(
        s1, 0, Wl1t, b1, 0, bias1, 0, B_, H_, T_);

    // h1 = xT @ W1c + bias1[b]            (M=N_, N=H, K=T, batched)
    k_gemm<false, false, false><<<dim3(H_ / 128, N_ / 128, B_), 256, 0, stream>>>(
        xT, (long)N_ * T_, W1ct, bias1, H_, h1, (long)N_ * H_, N_, H_, T_);

    k_colsum_h<<<(B_ * H_) / 256, 256, 0, stream>>>(s2, h1);

    // bias2[b,:] = s2[b,:] @ Wl2 + b2     (M=B, N=H, K=H, fp32 out)
    k_gemm<false, false, true><<<dim3(H_ / 128, 1, 1), 256, 0, stream>>>(
        s2, 0, Wl2t, b2, 0, bias2, 0, B_, H_, H_);

    // h2 = h1 @ W2c + bias2[b]            (M=N_, N=H, K=H, batched)
    k_gemm<false, false, false><<<dim3(H_ / 128, N_ / 128, B_), 256, 0, stream>>>(
        h1, (long)N_ * H_, W2ct, bias2, H_, h2, (long)N_ * H_, N_, H_, H_);

    // t1 = h2 @ W1a                        (reuse h1 buffer)
    __bf16* t1 = h1;
    k_gemm<false, false, false><<<dim3(H_ / 128, N_ / 128, B_), 256, 0, stream>>>(
        h2, (long)N_ * H_, W1at, nullptr, 0, t1, (long)N_ * H_, N_, H_, H_);

    // pT[b,t,n] = sigmoid(t1 @ W1b)[n,t]  (transposed store; reuse xT buffer)
    __bf16* pT = xT;
    k_gemm<true, true, false><<<dim3(T_ / 128, N_ / 128, B_), 256, 0, stream>>>(
        t1, (long)N_ * H_, W1bt, nullptr, 0, pT, (long)T_ * N_, N_, T_, H_);

    // t2 = pT @ W2a                        (M=T, N=N_, K=N_; reuse h2 buffer)
    __bf16* t2 = h2;
    k_gemm<false, false, false><<<dim3(N_ / 128, T_ / 128, B_), 256, 0, stream>>>(
        pT, (long)T_ * N_, W2at, nullptr, 0, t2, (long)T_ * N_, T_, N_, N_);

    // out = sigmoid(t2 @ W2b)              (M=T, N=H, K=N_, fp32 out)
    k_gemm<true, false, true><<<dim3(H_ / 128, T_ / 128, B_), 256, 0, stream>>>(
        t2, (long)T_ * N_, W2bt, nullptr, 0, (float*)d_out, (long)T_ * H_, T_, H_, N_);
}
